// PU_SAGE_31147102831272
// MI455X (gfx1250) — matrix operand order repaired and checked by reading, not compiled, not run on hardware
//
#include <hip/hip_runtime.h>
#include <hip/hip_bf16.h>
#include <math.h>

#define NN 50000
#define NE 800000
#define NG 512

typedef __attribute__((ext_vector_type(16))) __bf16 v16bf;
typedef __attribute__((ext_vector_type(8)))  __bf16 v8bf;
typedef __attribute__((ext_vector_type(8)))  float  v8f;

// ---------------- WMMA fragment loaders (ISA 7.12.2, wave32) ----------------

// A-matrix 16x32 bf16 (MxK): lanes 0-15 -> M=0..15, K {0-7,16-23};
// lanes 16-31 -> same M, K {8-15,24-31}. Two contiguous 16B chunks per lane.
static __device__ __forceinline__ v16bf load_a_frag(const __bf16* __restrict__ A,
                                                    int K, int mtile, int kk, int lane) {
  const int m    = mtile * 16 + (lane & 15);
  const int half = lane >> 4;
  const __bf16* row = A + (size_t)m * K + kk;
  v16bf a;
#pragma unroll
  for (int i = 0; i < 8; ++i) {
    const int kb = (i < 4) ? (8 * half + 2 * i) : (16 + 8 * half + 2 * (i - 4));
    a[2 * i]     = row[kb];
    a[2 * i + 1] = row[kb + 1];
  }
  return a;
}

// B-matrix 32x16 bf16 (KxN): lane%16 = column n, lanes 0-15 K {0-7,16-23}, lanes 16-31 K {8-15,24-31}.
// B[k][n] = W[n][k], W row-major [N x K] -> two contiguous 8-bf16 (16B) chunks per lane.
static __device__ __forceinline__ v16bf load_b_frag(const __bf16* __restrict__ W,
                                                    int K, int ntile, int kk, int lane) {
  const int n    = ntile * 16 + (lane & 15);
  const int half = lane >> 4;
  const __bf16* row = W + (size_t)n * K + kk;
  v16bf b;
#pragma unroll
  for (int i = 0; i < 16; ++i) b[i] = row[(i < 8) ? (8 * half + i) : (8 + 8 * half + i)];
  return b;
}

// ---- Fused dual-GEMM, 16x64 tile per wave (1 A-frag -> 4 B-frags/accums) ----
// out = act(Aagg@Wl^T + Ax@Wr^T + bias); if pool != nullptr, atomically
// accumulate activated rows into pool[batch[m]] (fused mean-pool numerator).
__global__ __launch_bounds__(32)
void wmma_gemm_dual(const __bf16* __restrict__ Aagg, const __bf16* __restrict__ Wl, int Kl,
                    const __bf16* __restrict__ Ax,   const __bf16* __restrict__ Wr, int Kr,
                    const float* __restrict__ bias, int N,
                    __bf16* __restrict__ outb,
                    float* __restrict__ pool, const int* __restrict__ batch,
                    int act /*0=leaky(0.01), 1=relu*/) {
  const int lane  = threadIdx.x & 31;
  const int mtile = blockIdx.x;
  const int nt0   = blockIdx.y * 4;      // first of 4 consecutive 16-wide N tiles
  v8f c0 = {0.f,0.f,0.f,0.f,0.f,0.f,0.f,0.f};
  v8f c1 = c0, c2 = c0, c3 = c0;

  for (int kk = 0; kk < Kl; kk += 32) {
    v16bf a  = load_a_frag(Aagg, Kl, mtile, kk, lane);
    v16bf b0 = load_b_frag(Wl, Kl, nt0 + 0, kk, lane);
    v16bf b1 = load_b_frag(Wl, Kl, nt0 + 1, kk, lane);
    v16bf b2 = load_b_frag(Wl, Kl, nt0 + 2, kk, lane);
    v16bf b3 = load_b_frag(Wl, Kl, nt0 + 3, kk, lane);
    c0 = __builtin_amdgcn_wmma_f32_16x16x32_bf16(false, a, false, b0, (short)0, c0, false, false);
    c1 = __builtin_amdgcn_wmma_f32_16x16x32_bf16(false, a, false, b1, (short)0, c1, false, false);
    c2 = __builtin_amdgcn_wmma_f32_16x16x32_bf16(false, a, false, b2, (short)0, c2, false, false);
    c3 = __builtin_amdgcn_wmma_f32_16x16x32_bf16(false, a, false, b3, (short)0, c3, false, false);
  }
  for (int kk = 0; kk < Kr; kk += 32) {
    v16bf a  = load_a_frag(Ax, Kr, mtile, kk, lane);
    v16bf b0 = load_b_frag(Wr, Kr, nt0 + 0, kk, lane);
    v16bf b1 = load_b_frag(Wr, Kr, nt0 + 1, kk, lane);
    v16bf b2 = load_b_frag(Wr, Kr, nt0 + 2, kk, lane);
    v16bf b3 = load_b_frag(Wr, Kr, nt0 + 3, kk, lane);
    c0 = __builtin_amdgcn_wmma_f32_16x16x32_bf16(false, a, false, b0, (short)0, c0, false, false);
    c1 = __builtin_amdgcn_wmma_f32_16x16x32_bf16(false, a, false, b1, (short)0, c1, false, false);
    c2 = __builtin_amdgcn_wmma_f32_16x16x32_bf16(false, a, false, b2, (short)0, c2, false, false);
    c3 = __builtin_amdgcn_wmma_f32_16x16x32_bf16(false, a, false, b3, (short)0, c3, false, false);
  }

  const int half = lane >> 4;
  v8f acc[4] = {c0, c1, c2, c3};
#pragma unroll
  for (int t = 0; t < 4; ++t) {
    const int n  = (nt0 + t) * 16 + (lane & 15);
    const float bn = bias[n];
#pragma unroll
    for (int j = 0; j < 8; ++j) {
      const int m = mtile * 16 + j + 8 * half;   // C/D layout: VGPR j -> M = j + 8*half
      float v = acc[t][j] + bn;
      v = (act == 0) ? (v > 0.f ? v : 0.01f * v) : fmaxf(v, 0.f);
      if (pool) {
        atomicAdd(&pool[(size_t)batch[m] * N + n], v);
      } else {
        outb[(size_t)m * N + n] = (__bf16)v;
      }
    }
  }
}

// ---------------- Graph / elementwise kernels ----------------

__global__ void k_deg(const int* __restrict__ ei, float* __restrict__ deg) {
  int e = blockIdx.x * blockDim.x + threadIdx.x;
  if (e >= NE) return;
  atomicAdd(&deg[ei[NE + e]], 1.0f);
}

__global__ void k_deginv(const float* __restrict__ deg, float* __restrict__ dinv) {
  int i = blockIdx.x * blockDim.x + threadIdx.x;
  if (i >= NN) return;
  dinv[i] = 1.0f / fmaxf(deg[i], 1.0f);
}

// f32 [N x Kin] -> bf16 [N x Kpad], zero padded in K
__global__ void k_cvt_pad(const float* __restrict__ src, __bf16* __restrict__ dst,
                          int Nrows, int Kin, int Kpad) {
  int t = blockIdx.x * blockDim.x + threadIdx.x;
  if (t >= Nrows * Kpad) return;
  int n = t / Kpad, k = t - n * Kpad;
  dst[t] = (k < Kin) ? (__bf16)src[(size_t)n * Kin + k] : (__bf16)0.0f;
}

// edge gather (bf16, 16B vector) + f32 atomic scatter-add; D multiple of 8
__global__ void k_agg(const __bf16* __restrict__ xb, const int* __restrict__ ei,
                      float* __restrict__ aggf, int D, int chunks /* = D/8 */) {
  int t = blockIdx.x * blockDim.x + threadIdx.x;
  int e = t / chunks;
  if (e >= NE) return;
  int f0 = (t - e * chunks) << 3;
  int src = ei[e];
  int dst = ei[NE + e];
  v8bf v = *(const v8bf*)(xb + (size_t)src * D + f0);
  float* o = aggf + (size_t)dst * D + f0;
#pragma unroll
  for (int i = 0; i < 8; ++i) atomicAdd(&o[i], (float)v[i]);
}

__global__ void k_scale_cvt(const float* __restrict__ aggf, const float* __restrict__ dinv,
                            __bf16* __restrict__ aggb, int D) {
  int t = blockIdx.x * blockDim.x + threadIdx.x;
  if (t >= NN * D) return;
  aggb[t] = (__bf16)(aggf[t] * dinv[t / D]);
}

__global__ void k_cnt(const int* __restrict__ batch, float* __restrict__ cnt) {
  int i = blockIdx.x * blockDim.x + threadIdx.x;
  if (i >= NN) return;
  atomicAdd(&cnt[batch[i]], 1.0f);
}

__global__ void k_pooldiv(const float* __restrict__ pool, const float* __restrict__ cnt,
                          __bf16* __restrict__ poolb) {
  int t = blockIdx.x * blockDim.x + threadIdx.x;
  if (t >= NG * 512) return;
  poolb[t] = (__bf16)(pool[t] / fmaxf(cnt[t >> 9], 1.0f));
}

__global__ void k_final(const __bf16* __restrict__ x5, const float* __restrict__ Wo,
                        const float* __restrict__ bo, float* __restrict__ out) {
  int g = blockIdx.x * blockDim.x + threadIdx.x;
  if (g >= NG) return;
  float s = bo[0];
#pragma unroll 4
  for (int k = 0; k < 128; ++k) s += (float)x5[(size_t)g * 128 + k] * Wo[k];
  out[g] = 1.0f / (1.0f + expf(-s));
}

// ---------------- Host launch ----------------

extern "C" void kernel_launch(void* const* d_in, const int* in_sizes, int n_in,
                              void* d_out, int out_size, void* d_ws, size_t ws_size,
                              hipStream_t stream) {
  (void)in_sizes; (void)n_in; (void)out_size; (void)ws_size;

  const float* x   = (const float*)d_in[0];
  const int*   ei  = (const int*)d_in[1];
  const int*   bat = (const int*)d_in[2];
  const float* Wl1 = (const float*)d_in[3];  const float* bl1 = (const float*)d_in[4];
  const float* Wr1 = (const float*)d_in[5];
  const float* Wl2 = (const float*)d_in[6];  const float* bl2 = (const float*)d_in[7];
  const float* Wr2 = (const float*)d_in[8];
  const float* Wl3 = (const float*)d_in[9];  const float* bl3 = (const float*)d_in[10];
  const float* Wr3 = (const float*)d_in[11];
  const float* Wf1 = (const float*)d_in[12]; const float* bf1 = (const float*)d_in[13];
  const float* Wf2 = (const float*)d_in[14]; const float* bf2 = (const float*)d_in[15];
  const float* Wo  = (const float*)d_in[16]; const float* bo  = (const float*)d_in[17];
  float* out = (float*)d_out;

  // workspace arena
  char* base = (char*)d_ws;
  size_t off = 0;
  auto alloc = [&](size_t bytes) -> char* {
    char* p = base + off;
    off = (off + bytes + 255) & ~(size_t)255;
    return p;
  };
  float*  deg   = (float*)alloc((size_t)NN * 4);
  float*  dinv  = (float*)alloc((size_t)NN * 4);
  float*  aggf  = (float*)alloc((size_t)NN * 256 * 4);
  __bf16* aggb  = (__bf16*)alloc((size_t)NN * 256 * 2);
  __bf16* x0b   = (__bf16*)alloc((size_t)NN * 64 * 2);
  __bf16* x1b   = (__bf16*)alloc((size_t)NN * 128 * 2);
  __bf16* x2b   = (__bf16*)alloc((size_t)NN * 256 * 2);
  __bf16* wl1b  = (__bf16*)alloc(128 * 64 * 2);
  __bf16* wr1b  = (__bf16*)alloc(128 * 64 * 2);
  __bf16* wl2b  = (__bf16*)alloc(256 * 128 * 2);
  __bf16* wr2b  = (__bf16*)alloc(256 * 128 * 2);
  __bf16* wl3b  = (__bf16*)alloc(512 * 256 * 2);
  __bf16* wr3b  = (__bf16*)alloc(512 * 256 * 2);
  __bf16* wf1b  = (__bf16*)alloc(256 * 512 * 2);
  __bf16* wf2b  = (__bf16*)alloc(128 * 256 * 2);
  float*  pool  = (float*)alloc((size_t)NG * 512 * 4);
  float*  cnt   = (float*)alloc((size_t)NG * 4);
  __bf16* poolb = (__bf16*)alloc((size_t)NG * 512 * 2);
  __bf16* x4b   = (__bf16*)alloc((size_t)NG * 256 * 2);
  __bf16* x5b   = (__bf16*)alloc((size_t)NG * 128 * 2);

  const int T = 256;
  auto cdiv = [](long long a, long long b) { return (unsigned)((a + b - 1) / b); };

  // ---- degree & counts ----
  hipMemsetAsync(deg, 0, (size_t)NN * 4, stream);
  hipMemsetAsync(pool, 0, (size_t)NG * 512 * 4, stream);
  hipMemsetAsync(cnt, 0, (size_t)NG * 4, stream);
  k_deg<<<cdiv(NE, T), T, 0, stream>>>(ei, deg);
  k_deginv<<<cdiv(NN, T), T, 0, stream>>>(deg, dinv);
  k_cnt<<<cdiv(NN, T), T, 0, stream>>>(bat, cnt);

  // ---- bf16 conversions (weights padded K: 50 -> 64) ----
  k_cvt_pad<<<cdiv((long long)NN * 64, T), T, 0, stream>>>(x, x0b, NN, 50, 64);
  k_cvt_pad<<<cdiv(128 * 64, T),  T, 0, stream>>>(Wl1, wl1b, 128, 50, 64);
  k_cvt_pad<<<cdiv(128 * 64, T),  T, 0, stream>>>(Wr1, wr1b, 128, 50, 64);
  k_cvt_pad<<<cdiv(256 * 128, T), T, 0, stream>>>(Wl2, wl2b, 256, 128, 128);
  k_cvt_pad<<<cdiv(256 * 128, T), T, 0, stream>>>(Wr2, wr2b, 256, 128, 128);
  k_cvt_pad<<<cdiv(512 * 256, T), T, 0, stream>>>(Wl3, wl3b, 512, 256, 256);
  k_cvt_pad<<<cdiv(512 * 256, T), T, 0, stream>>>(Wr3, wr3b, 512, 256, 256);
  k_cvt_pad<<<cdiv(256 * 512, T), T, 0, stream>>>(Wf1, wf1b, 256, 512, 512);
  k_cvt_pad<<<cdiv(128 * 256, T), T, 0, stream>>>(Wf2, wf2b, 128, 256, 256);

  const int MT = NN / 16;  // 3125 M-tiles (exact)

  // ---- layer 1: D=64 (padded from 50), N=128, leaky relu ----
  hipMemsetAsync(aggf, 0, (size_t)NN * 64 * 4, stream);
  k_agg<<<cdiv((long long)NE * 8, T), T, 0, stream>>>(x0b, ei, aggf, 64, 8);
  k_scale_cvt<<<cdiv((long long)NN * 64, T), T, 0, stream>>>(aggf, dinv, aggb, 64);
  wmma_gemm_dual<<<dim3(MT, 128 / 64), 32, 0, stream>>>(
      aggb, wl1b, 64, x0b, wr1b, 64, bl1, 128, x1b, nullptr, nullptr, 0);

  // ---- layer 2: D=128, N=256, relu ----
  hipMemsetAsync(aggf, 0, (size_t)NN * 128 * 4, stream);
  k_agg<<<cdiv((long long)NE * 16, T), T, 0, stream>>>(x1b, ei, aggf, 128, 16);
  k_scale_cvt<<<cdiv((long long)NN * 128, T), T, 0, stream>>>(aggf, dinv, aggb, 128);
  wmma_gemm_dual<<<dim3(MT, 256 / 64), 32, 0, stream>>>(
      aggb, wl2b, 128, x1b, wr2b, 128, bl2, 256, x2b, nullptr, nullptr, 1);

  // ---- layer 3: D=256, N=512, relu, fused mean-pool accumulate ----
  hipMemsetAsync(aggf, 0, (size_t)NN * 256 * 4, stream);
  k_agg<<<cdiv((long long)NE * 32, T), T, 0, stream>>>(x2b, ei, aggf, 256, 32);
  k_scale_cvt<<<cdiv((long long)NN * 256, T), T, 0, stream>>>(aggf, dinv, aggb, 256);
  wmma_gemm_dual<<<dim3(MT, 512 / 64), 32, 0, stream>>>(
      aggb, wl3b, 256, x2b, wr3b, 256, bl3, 512, nullptr, pool, bat, 1);

  // ---- mean pool -> MLP head ----
  k_pooldiv<<<cdiv((long long)NG * 512, T), T, 0, stream>>>(pool, cnt, poolb);
  wmma_gemm_dual<<<dim3(NG / 16, 256 / 64), 32, 0, stream>>>(
      poolb, wf1b, 512, nullptr, nullptr, 0, bf1, 256, x4b, nullptr, nullptr, 1);
  wmma_gemm_dual<<<dim3(NG / 16, 128 / 64), 32, 0, stream>>>(
      x4b, wf2b, 256, nullptr, nullptr, 0, bf2, 128, x5b, nullptr, nullptr, 1);
  k_final<<<cdiv(NG, T), T, 0, stream>>>(x5b, Wo, bo, out);
}
